// ScoreNetworkX_54107997995735
// MI455X (gfx1250) — matrix operand order repaired and checked by reading, not compiled, not run on hardware
//
#include <hip/hip_runtime.h>
#include <hip/hip_bf16.h>
#include <math.h>

// Problem constants (match reference)
#define BB 32
#define NN 128
#define FF 10
#define HIDD 64
#define NHIDD 32
#define CRANGE 7.5f      // 15.0 / NL_EGNN
#define EPS_R 1e-8f

typedef __attribute__((ext_vector_type(16))) _Float16 v16h;
typedef __attribute__((ext_vector_type(8)))  _Float16 v8h;
typedef __attribute__((ext_vector_type(8)))  float    v8f;
typedef __attribute__((ext_vector_type(4)))  unsigned int v4u;
typedef __attribute__((ext_vector_type(8)))  int      v8i;
typedef __attribute__((ext_vector_type(4)))  int      v4i;

union Frag16 { v16h v; v8h h[2]; _Float16 e[16]; };

__device__ __forceinline__ v8f vzero8f() {
    v8f v = {0.f,0.f,0.f,0.f,0.f,0.f,0.f,0.f};
    return v;
}
__device__ __forceinline__ float siluf(float x)    { return x / (1.f + __expf(-x)); }
__device__ __forceinline__ float sigmoidf_(float x){ return 1.f / (1.f + __expf(-x)); }
__device__ __forceinline__ float eluf(float x)     { return x > 0.f ? x : (__expf(x) - 1.f); }

// reuse_a/reuse_b must be literal constants -> macro with literals
#define WMMA_ACC(accv, Av, Bv, RA) \
    (accv) = __builtin_amdgcn_wmma_f32_16x16x32_f16( \
        false, (Av), false, (Bv), (short)0, (accv), (RA), false)

// Total packed weight set: wf1(10240) | wf2(4096) | wf3(1024) halves = 15360 halves
#define WSET_HALVES 15360

// ---------------------------------------------------------------------------
// Pack fp32 weight matrix W[K][Nout] (row-major, din x dout) into per-lane f16
// B-fragments: dst[(ks*ntiles+nt)*512 + lane*16 + v]
//   n = nt*16 + (lane&15); k = ks*32 + (lane>>4)*16 + v
// ---------------------------------------------------------------------------
__global__ void k_pack_b(const float* __restrict__ W, int K, int Nout,
                         int ksteps, int ntiles, _Float16* __restrict__ dst)
{
    int idx = blockIdx.x * blockDim.x + threadIdx.x;
    int total = ksteps * ntiles * 512;
    if (idx >= total) return;
    int f      = idx >> 9;
    int within = idx & 511;
    int lane = within >> 4, v = within & 15;
    int ks = f / ntiles, nt = f % ntiles;
    int n = nt * 16 + (lane & 15);
    int k = ks * 32 + ((lane >> 4) << 4) + v;
    float val = (k < K && n < Nout) ? W[k * Nout + n] : 0.f;
    dst[idx] = (_Float16)val;
}

// ---------------------------------------------------------------------------
// Fused per-edge MLP with WMMA. grid = B*8, block = 256 (8 waves).
// ---------------------------------------------------------------------------
__global__ __launch_bounds__(256) void k_edge_mlp(
    const _Float16* __restrict__ h16,   // (B,N,64) f16
    const float*    __restrict__ pos,   // (B,N,3)
    const float*    __restrict__ adj,   // (B,N,N)
    const _Float16* __restrict__ wset,  // packed weight set (WSET_HALVES halves)
    const float* b1, const float* b2, const float* b3,  // b3 null for coord
    float* __restrict__ out, int mode)
{
    __shared__ __align__(16) _Float16 sWall[WSET_HALVES]; // 30 KB contiguous
    __shared__ __align__(16) _Float16 ylds[8 * 16 * 64];  // 16 KB (wave-private slabs)
    __shared__ float sAgg[16 * 64];                       //  4 KB
    __shared__ float sRow[8 * 16];
    __shared__ float sB1[64], sB2[64];

    const int tid  = threadIdx.x;
    const int lane = tid & 31;
    const int wave = tid >> 5;
    const int bidx = blockIdx.x >> 3;
    const int i0   = (blockIdx.x & 7) << 4;

    // ---- Stage packed weights into LDS via the Tensor Data Mover ----
#if __has_builtin(__builtin_amdgcn_tensor_load_to_lds)
    if (tid < 32) {   // wave 0 issues one TDM descriptor; EXEC is ignored by TDM
        unsigned long long ga = (unsigned long long)(uintptr_t)wset;
        unsigned ldsoff = (unsigned)(uintptr_t)(void*)sWall;  // LDS aperture: low 32 bits
        v4u g0;
        g0.x = 1u;                                          // count=1, user mode
        g0.y = ldsoff;                                      // lds_addr
        g0.z = (unsigned)(ga & 0xFFFFFFFFull);              // global_addr[31:0]
        g0.w = (unsigned)((ga >> 32) & 0x1FFFFFFull) | (2u << 30); // addr[56:32] | type=2
        v8i g1;
        g1[0] = 0x00010000;                                 // data_size=2B, no multicast
        g1[1] = (int)((WSET_HALVES & 0xFFFF) << 16);        // tensor_dim0[15:0] @ bit48
        g1[2] = (int)((WSET_HALVES >> 16) | (1 << 16));     // tensor_dim0 hi | tensor_dim1=1
        g1[3] = (int)((unsigned)WSET_HALVES << 16);         // tile_dim0 @ bit112
        g1[4] = 1;                                          // tile_dim1=1
        g1[5] = WSET_HALVES;                                // tensor_dim0_stride lo
        g1[6] = 0; g1[7] = 0;
        v4i z4 = {0, 0, 0, 0};
#if __clang_major__ >= 23
        v8i z8 = {0, 0, 0, 0, 0, 0, 0, 0};
        __builtin_amdgcn_tensor_load_to_lds(g0, g1, z4, z4, z8, 0);
#else
        __builtin_amdgcn_tensor_load_to_lds(g0, g1, z4, z4, 0);
#endif
        __builtin_amdgcn_s_wait_tensorcnt(0);
    }
#else
    {
        const uint4* src = (const uint4*)wset;
        uint4* d1 = (uint4*)sWall;                          // 1920 uint4 total
        for (int k = tid; k < 1920; k += 256) d1[k] = src[k];
    }
#endif
    if (tid < 64) { sB1[tid] = b1[tid]; sB2[tid] = b2[tid]; }
    for (int k = tid; k < 16 * 64; k += 256) sAgg[k] = 0.f;
    __syncthreads();

    const _Float16* sW1 = sWall;
    const _Float16* sW2 = sWall + 10240;
    const _Float16* sW3 = sWall + 14336;

    const int   colL   = lane & 15;
    const int   hi8    = (lane >> 4) * 16;
    const int   rowoff = (lane >> 4) * 8;
    const int   irow   = i0 + colL;                       // A-row for h_i side
    const _Float16* hrow_i = h16 + ((size_t)(bidx * NN + irow)) * HIDD;
    const float pix = pos[(bidx * NN + irow) * 3 + 0];
    const float piy = pos[(bidx * NN + irow) * 3 + 1];
    const float piz = pos[(bidx * NN + irow) * 3 + 2];
    const float b3v = (b3 != nullptr) ? b3[0] : 0.f;
    _Float16* yw = ylds + wave * 16 * 64;
    const _Float16* yrow = yw + colL * 64;

    // Hoist loop-invariant B-fragments for GEMM2 (8) and GEMM3 (2) into VGPRs
    Frag16 w2f[8], w3f[2];
#pragma unroll
    for (int f = 0; f < 8; ++f) {
        const _Float16* p = sW2 + f * 512 + lane * 16;
        w2f[f].h[0] = *(const v8h*)p;
        w2f[f].h[1] = *(const v8h*)(p + 8);
    }
#pragma unroll
    for (int f = 0; f < 2; ++f) {
        const _Float16* p = sW3 + f * 512 + lane * 16;
        w3f[f].h[0] = *(const v8h*)p;
        w3f[f].h[1] = *(const v8h*)(p + 8);
    }

    for (int jc = 0; jc < NN; jc += 8) {
        const int j = jc + wave;
        const _Float16* hrow_j = h16 + ((size_t)(bidx * NN + j)) * HIDD;
        __builtin_prefetch(hrow_j + 8 * HIDD, 0, 1);      // next j-chunk row

        const float pjx = pos[(bidx * NN + j) * 3 + 0];
        const float pjy = pos[(bidx * NN + j) * 3 + 1];
        const float pjz = pos[(bidx * NN + j) * 3 + 2];
        const float dx = pix - pjx, dy = piy - pjy, dz = piz - pjz;
        const float radial = dx * dx + dy * dy + dz * dz;
        const float adjv = adj[((size_t)(bidx * NN + irow)) * NN + j];

        // ---- GEMM1: X(16x160) @ W1(160x64) ----
        v8f acc[4];
#pragma unroll
        for (int nt = 0; nt < 4; ++nt) acc[nt] = vzero8f();
#pragma unroll
        for (int ks = 0; ks < 5; ++ks) {
            Frag16 a;
            if (ks < 2) {
                a.h[0] = *(const v8h*)(hrow_i + ks * 32 + hi8);
                a.h[1] = *(const v8h*)(hrow_i + ks * 32 + 8 + hi8);
            } else if (ks < 4) {
                a.h[0] = *(const v8h*)(hrow_j + (ks - 2) * 32 + hi8);
                a.h[1] = *(const v8h*)(hrow_j + (ks - 2) * 32 + 8 + hi8);
            } else {
#pragma unroll
                for (int q = 0; q < 16; ++q) a.e[q] = (_Float16)0.f;
                if (lane < 16) { a.e[0] = (_Float16)radial; a.e[1] = (_Float16)adjv; }
            }
            Frag16 bf0, bf1, bf2, bf3;
            {
                const _Float16* p0 = sW1 + (ks * 4 + 0) * 512 + lane * 16;
                const _Float16* p1 = sW1 + (ks * 4 + 1) * 512 + lane * 16;
                const _Float16* p2 = sW1 + (ks * 4 + 2) * 512 + lane * 16;
                const _Float16* p3 = sW1 + (ks * 4 + 3) * 512 + lane * 16;
                bf0.h[0] = *(const v8h*)p0; bf0.h[1] = *(const v8h*)(p0 + 8);
                bf1.h[0] = *(const v8h*)p1; bf1.h[1] = *(const v8h*)(p1 + 8);
                bf2.h[0] = *(const v8h*)p2; bf2.h[1] = *(const v8h*)(p2 + 8);
                bf3.h[0] = *(const v8h*)p3; bf3.h[1] = *(const v8h*)(p3 + 8);
            }
            // reuse-A hint: same A operand feeds 4 consecutive identical WMMAs
            WMMA_ACC(acc[0], a.v, bf0.v, false);
            WMMA_ACC(acc[1], a.v, bf1.v, true);
            WMMA_ACC(acc[2], a.v, bf2.v, true);
            WMMA_ACC(acc[3], a.v, bf3.v, true);
        }
        // bias + silu -> f16 staging (C layout -> row-major in LDS)
#pragma unroll
        for (int nt = 0; nt < 4; ++nt)
#pragma unroll
            for (int r = 0; r < 8; ++r) {
                float v = acc[nt][r] + sB1[nt * 16 + colL];
                v = siluf(v);
                yw[(rowoff + r) * 64 + nt * 16 + colL] = (_Float16)v;
            }
        asm volatile("s_wait_dscnt 0x0" ::: "memory");

        // ---- GEMM2: Y1(16x64) @ W2(64x64), B-frags in registers ----
        v8f acc2[4];
#pragma unroll
        for (int nt = 0; nt < 4; ++nt) acc2[nt] = vzero8f();
#pragma unroll
        for (int ks = 0; ks < 2; ++ks) {
            Frag16 a;
            a.h[0] = *(const v8h*)(yrow + ks * 32 + hi8);
            a.h[1] = *(const v8h*)(yrow + ks * 32 + 8 + hi8);
            WMMA_ACC(acc2[0], a.v, w2f[ks * 4 + 0].v, false);
            WMMA_ACC(acc2[1], a.v, w2f[ks * 4 + 1].v, true);
            WMMA_ACC(acc2[2], a.v, w2f[ks * 4 + 2].v, true);
            WMMA_ACC(acc2[3], a.v, w2f[ks * 4 + 3].v, true);
        }
        float y2v[4][8];
#pragma unroll
        for (int nt = 0; nt < 4; ++nt)
#pragma unroll
            for (int r = 0; r < 8; ++r) {
                float v = acc2[nt][r] + sB2[nt * 16 + colL];
                v = siluf(v);
                y2v[nt][r] = v;
                yw[(rowoff + r) * 64 + nt * 16 + colL] = (_Float16)v;
            }
        asm volatile("s_wait_dscnt 0x0" ::: "memory");

        // ---- GEMM3: Y2(16x64) @ W3(64x1 padded to 16), B-frags in registers ----
        v8f acc3 = vzero8f();
#pragma unroll
        for (int ks = 0; ks < 2; ++ks) {
            Frag16 a;
            a.h[0] = *(const v8h*)(yrow + ks * 32 + hi8);
            a.h[1] = *(const v8h*)(yrow + ks * 32 + 8 + hi8);
            WMMA_ACC(acc3, a.v, w3f[ks].v, false);
        }
        if (colL == 0) {
#pragma unroll
            for (int r = 0; r < 8; ++r) sRow[wave * 16 + rowoff + r] = acc3[r];
        }
        asm volatile("s_wait_dscnt 0x0" ::: "memory");

        if (mode == 0) {
#pragma unroll
            for (int r = 0; r < 8; ++r) {
                const int row = rowoff + r;
                float att = sigmoidf_(sRow[wave * 16 + row] + b3v);
                float em  = (i0 + row != j) ? 1.f : 0.f;
                float ae  = att * em;
#pragma unroll
                for (int nt = 0; nt < 4; ++nt)
                    atomicAdd(&sAgg[row * 64 + nt * 16 + colL], y2v[nt][r] * ae);
            }
        } else {
            if (lane < 16) {
                const int row = lane;
                const int i = i0 + row;
                float s  = sRow[wave * 16 + row];
                float th = tanhf(s) * CRANGE;
                float em = (i != j) ? 1.f : 0.f;
                float ddx = pos[(bidx * NN + i) * 3 + 0] - pjx;
                float ddy = pos[(bidx * NN + i) * 3 + 1] - pjy;
                float ddz = pos[(bidx * NN + i) * 3 + 2] - pjz;
                float rad = ddx * ddx + ddy * ddy + ddz * ddz;
                float sc  = th * em / (sqrtf(rad + EPS_R) + 1.f);
                atomicAdd(&sAgg[row * 4 + 0], ddx * sc);
                atomicAdd(&sAgg[row * 4 + 1], ddy * sc);
                atomicAdd(&sAgg[row * 4 + 2], ddz * sc);
            }
        }
    }
    __syncthreads();

    if (mode == 0) {
        for (int k = tid; k < 16 * 64; k += 256) {
            int row = k >> 6, c = k & 63;
            out[((size_t)(bidx * NN + i0 + row)) * 64 + c] = sAgg[k];
        }
    } else {
        if (tid < 48) {
            int row = tid / 3, d = tid % 3;
            out[((size_t)(bidx * NN + i0 + row)) * 3 + d] = sAgg[row * 4 + d];
        }
    }
}

// ---------------------------------------------------------------------------
// Generic node-wise linear: out = act(in0@W[:w0] + in1@W[w0:] + b) (+res)(*mask)
// act: 0=none 1=silu 2=tanh 3=elu
// ---------------------------------------------------------------------------
__global__ void k_node_linear(const float* in0, int w0, int s0,
                              const float* in1, int w1, int s1,
                              const float* W, const float* bias,
                              const float* residual, const float* mask,
                              float* out, int wout, int sout,
                              int act, int nodes)
{
    int idx = blockIdx.x * blockDim.x + threadIdx.x;
    if (idx >= nodes * wout) return;
    int node = idx / wout, c = idx % wout;
    float s = bias ? bias[c] : 0.f;
    const float* r0 = in0 + (size_t)node * s0;
    for (int k = 0; k < w0; ++k) s += r0[k] * W[k * wout + c];
    if (in1) {
        const float* r1 = in1 + (size_t)node * s1;
        for (int k = 0; k < w1; ++k) s += r1[k] * W[(w0 + k) * wout + c];
    }
    if (act == 1) s = siluf(s);
    else if (act == 2) s = tanhf(s);
    else if (act == 3) s = eluf(s);
    if (residual) s += residual[(size_t)node * wout + c];
    if (mask) s *= mask[node];
    out[(size_t)node * sout + c] = s;
}

__global__ void k_f32_to_f16(const float* __restrict__ in, _Float16* __restrict__ out, int n)
{
    int idx = blockIdx.x * blockDim.x + threadIdx.x;
    if (idx < n) out[idx] = (_Float16)in[idx];
}

__global__ void k_scatter(const float* in, int w, int sin,
                          float* out, int sout, int act, int nodes)
{
    int idx = blockIdx.x * blockDim.x + threadIdx.x;
    if (idx >= nodes * w) return;
    int node = idx / w, c = idx % w;
    float v = in[(size_t)node * sin + c];
    if (act == 2) v = tanhf(v);
    out[(size_t)node * sout + c] = v;
}

__global__ void k_build_h0(const float* __restrict__ x, const float* __restrict__ t,
                           float* __restrict__ out)
{
    int idx = blockIdx.x * blockDim.x + threadIdx.x;
    if (idx >= BB * NN * 11) return;
    int node = idx / 11, c = idx % 11;
    out[idx] = (c < FF) ? x[(size_t)node * FF + c] : t[node / NN];
}

__global__ void k_pos_add(float* posw, const float* aggP, const float* flags, int n)
{
    int idx = blockIdx.x * blockDim.x + threadIdx.x;
    if (idx >= n) return;
    posw[idx] = (posw[idx] + aggP[idx]) * flags[idx / 3];
}

__global__ void k_pos_center(const float* posw, const float* pos_in,
                             const float* flags, float* out_a, float* out_b)
{
    __shared__ float sd[NN * 3];
    __shared__ float ssum[4];
    int b = blockIdx.x, i = threadIdx.x;
    int node = b * NN + i;
    float m = flags[node];
    float d[3];
    for (int q = 0; q < 3; ++q) {
        d[q] = (posw[node * 3 + q] - pos_in[node * 3 + q]) * m;
        sd[i * 3 + q] = d[q];
    }
    __syncthreads();
    if (i < 4) {
        float s = 0.f;
        for (int k = 0; k < NN; ++k)
            s += (i < 3) ? sd[k * 3 + i] : flags[b * NN + k];
        ssum[i] = s;
    }
    __syncthreads();
    for (int q = 0; q < 3; ++q) {
        float v = d[q] - (ssum[q] / ssum[3]) * m;
        out_a[node * 3 + q] = v;
        out_b[node * 3 + q] = v;
    }
}

__global__ void k_copyf(const float* in, float* out, int n)
{
    int idx = blockIdx.x * blockDim.x + threadIdx.x;
    if (idx < n) out[idx] = in[idx];
}

// ---------------------------------------------------------------------------
extern "C" void kernel_launch(void* const* d_in, const int* in_sizes, int n_in,
                              void* d_out, int out_size, void* d_ws, size_t ws_size,
                              hipStream_t stream)
{
    (void)in_sizes; (void)n_in; (void)out_size; (void)ws_size;
    const float* x     = (const float*)d_in[0];
    const float* pos   = (const float*)d_in[1];
    const float* adj   = (const float*)d_in[2];
    const float* flags = (const float*)d_in[3];
    const float* t     = (const float*)d_in[4];

    // Params flattened in jax pytree order (dict keys sorted):
    // final[0..2]{W,b}, layers[0..1]{ blocks[0..1]{ coord{c1{W,b},c2{W,b},c3{W}},
    //   gcls[0]{att{W,b},e1{W,b},e2{W,b},n1{W,b},n2{W,b}} }, emb{W,b}, emb_out{W,b} }
    int pi = 5;
    auto nxt = [&]() { return (const float*)d_in[pi++]; };
    const float *fW1 = nxt(), *fb1 = nxt(), *fW2 = nxt(), *fb2 = nxt(), *fW3 = nxt(), *fb3 = nxt();
    struct Gcl   { const float *attW,*attb,*e1W,*e1b,*e2W,*e2b,*n1W,*n1b,*n2W,*n2b; };
    struct Coord { const float *c1W,*c1b,*c2W,*c2b,*c3W; };
    struct Blk   { Coord c; Gcl g; };
    struct Layer { Blk blk[2]; const float *embW,*embb,*outW,*outb; };
    Layer L[2];
    for (int l = 0; l < 2; ++l) {
        for (int k = 0; k < 2; ++k) {
            Blk& B_ = L[l].blk[k];
            B_.c.c1W = nxt(); B_.c.c1b = nxt(); B_.c.c2W = nxt(); B_.c.c2b = nxt(); B_.c.c3W = nxt();
            B_.g.attW = nxt(); B_.g.attb = nxt();
            B_.g.e1W  = nxt(); B_.g.e1b  = nxt();
            B_.g.e2W  = nxt(); B_.g.e2b  = nxt();
            B_.g.n1W  = nxt(); B_.g.n1b  = nxt();
            B_.g.n2W  = nxt(); B_.g.n2b  = nxt();
        }
        L[l].embW = nxt(); L[l].embb = nxt(); L[l].outW = nxt(); L[l].outb = nxt();
    }

    const int BN = BB * NN;
    char* wsp = (char*)d_ws;
    auto alloc = [&](size_t bytes) -> void* {
        void* p = (void*)wsp;
        wsp += (bytes + 1023) & ~(size_t)1023;
        return p;
    };
    float*     h      = (float*)alloc((size_t)BN * 64 * 4);
    _Float16*  h16    = (_Float16*)alloc((size_t)BN * 64 * 2);
    float*     agg    = (float*)alloc((size_t)BN * 64 * 4);
    float*     tmpn   = (float*)alloc((size_t)BN * 64 * 4);
    float*     h0     = (float*)alloc((size_t)BN * 11 * 4);
    float*     xs     = (float*)alloc((size_t)BN * 74 * 4);
    float*     f1b    = (float*)alloc((size_t)BN * 148 * 4);
    float*     f2b    = (float*)alloc((size_t)BN * 148 * 4);
    float*     pos_in = (float*)alloc((size_t)BN * 3 * 4);
    float*     posw   = (float*)alloc((size_t)BN * 3 * 4);
    float*     aggP   = (float*)alloc((size_t)BN * 3 * 4);
    float*     t32    = (float*)alloc((size_t)BN * 32 * 4);
    _Float16*  packs  = (_Float16*)alloc((size_t)8 * WSET_HALVES * 2);

    auto cdiv = [](int a, int b) { return (a + b - 1) / b; };
    auto pack = [&](const float* W, int K, int Nout, int ks, int nt, _Float16* dst) {
        int total = ks * nt * 512;
        k_pack_b<<<cdiv(total, 256), 256, 0, stream>>>(W, K, Nout, ks, nt, dst);
    };

    // Pre-pack all WMMA weight sets
    for (int l = 0; l < 2; ++l)
        for (int k = 0; k < 2; ++k) {
            const Blk& B_ = L[l].blk[k];
            _Float16* g = packs + ((size_t)((l * 2 + k) * 2 + 0)) * WSET_HALVES;
            pack(B_.g.e1W, 130, 64, 5, 4, g);
            pack(B_.g.e2W,  64, 64, 2, 4, g + 10240);
            pack(B_.g.attW, 64,  1, 2, 1, g + 14336);
            _Float16* c = packs + ((size_t)((l * 2 + k) * 2 + 1)) * WSET_HALVES;
            pack(B_.c.c1W, 130, 64, 5, 4, c);
            pack(B_.c.c2W,  64, 64, 2, 4, c + 10240);
            pack(B_.c.c3W,  64,  1, 2, 1, c + 14336);
        }

    // Init
    k_copyf<<<cdiv(BN * 3, 256), 256, 0, stream>>>(pos, pos_in, BN * 3);
    k_copyf<<<cdiv(BN * 3, 256), 256, 0, stream>>>(pos, posw, BN * 3);
    k_build_h0<<<cdiv(BN * 11, 256), 256, 0, stream>>>(x, t, h0);
    k_scatter<<<cdiv(BN * FF, 256), 256, 0, stream>>>(x, FF, FF, xs, 74, 0, BN);

    for (int l = 0; l < 2; ++l) {
        // emb
        if (l == 0)
            k_node_linear<<<cdiv(BN * 64, 256), 256, 0, stream>>>(
                h0, 11, 11, nullptr, 0, 0, L[l].embW, L[l].embb,
                nullptr, nullptr, h, 64, 64, 0, BN);
        else
            k_node_linear<<<cdiv(BN * 64, 256), 256, 0, stream>>>(
                xs + 10, 32, 74, nullptr, 0, 0, L[l].embW, L[l].embb,
                nullptr, nullptr, h, 64, 64, 0, BN);

        for (int k = 0; k < 2; ++k) {
            const Blk& B_ = L[l].blk[k];
            _Float16* gset = packs + ((size_t)((l * 2 + k) * 2 + 0)) * WSET_HALVES;
            _Float16* cset = packs + ((size_t)((l * 2 + k) * 2 + 1)) * WSET_HALVES;

            // GCL
            k_f32_to_f16<<<cdiv(BN * 64, 256), 256, 0, stream>>>(h, h16, BN * 64);
            k_edge_mlp<<<BB * 8, 256, 0, stream>>>(h16, posw, adj, gset,
                                                   B_.g.e1b, B_.g.e2b, B_.g.attb, agg, 0);
            k_node_linear<<<cdiv(BN * 64, 256), 256, 0, stream>>>(
                h, 64, 64, agg, 64, 64, B_.g.n1W, B_.g.n1b,
                nullptr, nullptr, tmpn, 64, 64, 1, BN);
            k_node_linear<<<cdiv(BN * 64, 256), 256, 0, stream>>>(
                tmpn, 64, 64, nullptr, 0, 0, B_.g.n2W, B_.g.n2b,
                h, flags, h, 64, 64, 0, BN);

            // Coord update (uses updated h, same pos)
            k_f32_to_f16<<<cdiv(BN * 64, 256), 256, 0, stream>>>(h, h16, BN * 64);
            k_edge_mlp<<<BB * 8, 256, 0, stream>>>(h16, posw, adj, cset,
                                                   B_.c.c1b, B_.c.c2b, nullptr, aggP, 1);
            k_pos_add<<<cdiv(BN * 3, 256), 256, 0, stream>>>(posw, aggP, flags, BN * 3);
        }

        // emb_out (+mask), tanh -> xs slice (also next layer's h source)
        k_node_linear<<<cdiv(BN * 32, 256), 256, 0, stream>>>(
            h, 64, 64, nullptr, 0, 0, L[l].outW, L[l].outb,
            nullptr, flags, t32, 32, 32, 0, BN);
        k_scatter<<<cdiv(BN * 32, 256), 256, 0, stream>>>(
            t32, 32, 32, xs + 10 + l * 32, 74, 2, BN);

        // pos = centered((posw - pos_in)*mask); becomes start pos of next layer
        k_pos_center<<<BB, NN, 0, stream>>>(posw, pos_in, flags, pos_in, posw);
    }

    // Final head: 74 -> 148 (elu) -> 148 (elu) -> 10 (*mask)
    k_node_linear<<<cdiv(BN * 148, 256), 256, 0, stream>>>(
        xs, 74, 74, nullptr, 0, 0, fW1, fb1, nullptr, nullptr, f1b, 148, 148, 3, BN);
    k_node_linear<<<cdiv(BN * 148, 256), 256, 0, stream>>>(
        f1b, 148, 148, nullptr, 0, 0, fW2, fb2, nullptr, nullptr, f2b, 148, 148, 3, BN);
    k_node_linear<<<cdiv(BN * FF, 256), 256, 0, stream>>>(
        f2b, 148, 148, nullptr, 0, 0, fW3, fb3, nullptr, flags,
        (float*)d_out, FF, FF, 0, BN);
}
